// DynamicConvEncoderLayer_60790967107733
// MI455X (gfx1250) — compile-verified, matrix operand order checked by reading, not run on hardware
//
#include <hip/hip_runtime.h>
#include <hip/hip_bf16.h>

typedef __attribute__((ext_vector_type(16))) _Float16 v16h;
typedef __attribute__((ext_vector_type(8)))  float    v8f;
typedef __attribute__((ext_vector_type(4)))  _Float16 half4_t;
typedef __attribute__((ext_vector_type(4)))  float    float4_t;

#define TDIM 4096
#define BDIM 16
#define EDIM 512
#define MTOT (TDIM * BDIM)   // 65536 rows
#define KD   512             // reduction dim for all GEMMs
#define KBLK (KD / 32)       // 16 k-steps of 32
#define HEADS 8
#define KW    31
#define PADL  15
#define NWPAD 256            // H*K = 248 padded to 256

// ---------------------------------------------------------------------------
// WMMA fragment-order addressing (halfs), per CDNA5 ISA 7.12.2.
// A-matrix (M x K, 16-bit): lane = (m&15) + 16*((k>>3)&1),
//   half index i = (k&7) + 8*((k>>4)&1), tiles of 16 rows x 32 k.
// ---------------------------------------------------------------------------
__device__ __forceinline__ int afrag_idx(int m, int k) {
  return ((((m >> 4) * KBLK + (k >> 5)) * 32) +
          ((m & 15) + (((k >> 3) & 1) << 4))) * 16 +
         ((k & 7) + (((k >> 4) & 1) << 3));
}
// B-matrix (K x N, 16-bit): same pattern with lane keyed by column n.
__device__ __forceinline__ int bfrag_idx(int k, int n) {
  return ((((n >> 4) * KBLK + (k >> 5)) * 32) +
          ((n & 15) + (((k >> 3) & 1) << 4))) * 16 +
         ((k & 7) + (((k >> 4) & 1) << 3));
}

// ---------------------------------------------------------------------------
// Prep: x [M,512] f32 -> A-fragment f16
// ---------------------------------------------------------------------------
__global__ void cvt_x_frag(const float* __restrict__ x, _Float16* __restrict__ xf) {
  int idx = blockIdx.x * blockDim.x + threadIdx.x;   // MTOT*128 threads
  int m = idx >> 7;
  int k = (idx & 127) << 2;
  float4_t v = *(const float4_t*)(x + m * KD + k);
  half4_t h;
  h[0] = (_Float16)v[0]; h[1] = (_Float16)v[1];
  h[2] = (_Float16)v[2]; h[3] = (_Float16)v[3];
  *(half4_t*)(xf + afrag_idx(m, k)) = h;             // i in {0,4,8,12}: contiguous, 8B aligned
}

// ---------------------------------------------------------------------------
// Prep: weight [512, nsrc] f32 -> B-fragment f16, columns padded to npad with 0
// launch with npad*128 threads
// ---------------------------------------------------------------------------
__global__ void cvt_w_frag(const float* __restrict__ w, _Float16* __restrict__ wf, int nsrc) {
  int idx = blockIdx.x * blockDim.x + threadIdx.x;
  int n = idx >> 7;
  int k = (idx & 127) << 2;
  half4_t h;
#pragma unroll
  for (int j = 0; j < 4; ++j) {
    float v = (n < nsrc) ? w[(k + j) * nsrc + n] : 0.0f;
    h[j] = (_Float16)v;
  }
  *(half4_t*)(wf + bfrag_idx(k, n)) = h;
}

// ---------------------------------------------------------------------------
// GEMM (f16 WMMA, f32 accumulate). Block = 128 threads (4 waves), tile 64Mx128N,
// wave tile 64x32 (4x2 16x16 accum tiles), K-loop over 16 steps of 32.
// MODE 0: D = mask ? 0 : (A@B + bias)  -> f16 fragment layout (h)
// MODE 1: D = A@B + bias[n<nsrc]       -> f32 plain [M, NWPAD] (logits)
// ---------------------------------------------------------------------------
template <int MODE>
__global__ void gemm_wmma(const _Float16* __restrict__ A,
                          const _Float16* __restrict__ Bw,
                          const float* __restrict__ bias,
                          const unsigned char* __restrict__ mask,
                          void* __restrict__ out, int nsrc) {
  const int lane = threadIdx.x & 31;
  const int wv   = threadIdx.x >> 5;
  const int m0   = blockIdx.y * 64;
  const int n0   = blockIdx.x * 128 + wv * 32;
  const int mt0  = m0 >> 4;
  const int nt0  = n0 >> 4;

  v8f acc[4][2] = {};

  for (int kb = 0; kb < KBLK; ++kb) {
    v16h a[4], b[2];
#pragma unroll
    for (int mt = 0; mt < 4; ++mt)
      a[mt] = *(const v16h*)(A + ((((mt0 + mt) * KBLK + kb) * 32) + lane) * 16);
#pragma unroll
    for (int nt = 0; nt < 2; ++nt)
      b[nt] = *(const v16h*)(Bw + ((((nt0 + nt) * KBLK + kb) * 32) + lane) * 16);
#pragma unroll
    for (int mt = 0; mt < 4; ++mt)
#pragma unroll
      for (int nt = 0; nt < 2; ++nt)
        acc[mt][nt] = __builtin_amdgcn_wmma_f32_16x16x32_f16(
            false, a[mt], false, b[nt], (short)0, acc[mt][nt], false, false);
  }

  // Epilogue. C/D layout: VGPR v holds row v (+8 for lanes 16-31), col = lane&15.
  const int rbase = (lane >> 4) << 3;
  const int cl    = lane & 15;
#pragma unroll
  for (int mt = 0; mt < 4; ++mt) {
#pragma unroll
    for (int nt = 0; nt < 2; ++nt) {
      const int n = n0 + nt * 16 + cl;
#pragma unroll
      for (int v = 0; v < 8; ++v) {
        const int m = m0 + mt * 16 + rbase + v;
        float val = acc[mt][nt][v];
        if (MODE == 0) {
          val += bias[n];
          const int t  = m >> 4;       // m = t*B + b, B = 16
          const int bb = m & 15;
          if (mask[bb * TDIM + t]) val = 0.0f;
          ((_Float16*)out)[afrag_idx(m, n)] = (_Float16)val;
        } else {
          val += (n < nsrc) ? bias[n] : 0.0f;
          ((float*)out)[m * NWPAD + n] = val;
        }
      }
    }
  }
}

// ---------------------------------------------------------------------------
// Softmax over the 31 taps of each (row, head). One thread per (m, head).
// ---------------------------------------------------------------------------
__global__ void softmax31(float* __restrict__ logits) {
  int idx = blockIdx.x * blockDim.x + threadIdx.x;   // MTOT*HEADS threads
  int m = idx >> 3;
  int hh = idx & 7;
  float* p = logits + m * NWPAD + hh * KW;
  float v[KW];
  float mx = -1e30f;
#pragma unroll
  for (int k = 0; k < KW; ++k) { v[k] = p[k]; mx = fmaxf(mx, v[k]); }
  float s = 0.0f;
#pragma unroll
  for (int k = 0; k < KW; ++k) { v[k] = __expf(v[k] - mx); s += v[k]; }
  float inv = 1.0f / s;
#pragma unroll
  for (int k = 0; k < KW; ++k) p[k] = v[k] * inv;
}

// ---------------------------------------------------------------------------
// Dynamic depthwise conv: out[t,b,c] = sum_k w[t,b,head,k] * h[t+k-15, b, c].
// One thread per (m, 4 channels); h is L2-resident f16 in fragment layout.
// ---------------------------------------------------------------------------
__global__ void dynconv(const _Float16* __restrict__ hf,
                        const float* __restrict__ w,
                        _Float16* __restrict__ of) {
  int idx = blockIdx.x * blockDim.x + threadIdx.x;   // MTOT*128 threads
  int m = idx >> 7;
  int c = (idx & 127) << 2;
  int t  = m >> 4;
  int bb = m & 15;
  int head = c >> 6;                                  // R = 64 channels per head
  const float* wp = w + m * NWPAD + head * KW;
  float a0 = 0.f, a1 = 0.f, a2 = 0.f, a3 = 0.f;
#pragma unroll
  for (int k = 0; k < KW; ++k) {
    int tt = t + k - PADL;
    if ((unsigned)tt < (unsigned)TDIM) {
      half4_t hv = *(const half4_t*)(hf + afrag_idx((tt << 4) | bb, c));
      float wk = wp[k];
      a0 += wk * (float)hv[0]; a1 += wk * (float)hv[1];
      a2 += wk * (float)hv[2]; a3 += wk * (float)hv[3];
    }
  }
  half4_t o;
  o[0] = (_Float16)a0; o[1] = (_Float16)a1; o[2] = (_Float16)a2; o[3] = (_Float16)a3;
  *(half4_t*)(of + afrag_idx(m, c)) = o;
}

// ---------------------------------------------------------------------------
// GEMM2 + bias + residual + LayerNorm fused. Block = 128 threads (4 waves),
// tile = 16 rows x full 512 cols (each wave: 8 col-tiles, 8 accum tiles).
// ---------------------------------------------------------------------------
__global__ void gemm2_ln(const _Float16* __restrict__ A,
                         const _Float16* __restrict__ Bw,
                         const float* __restrict__ b2,
                         const float* __restrict__ xres,
                         const float* __restrict__ gamma,
                         const float* __restrict__ beta,
                         float* __restrict__ out) {
  __shared__ float ylds[16 * 512];
  __shared__ float red1[16 * 8];
  __shared__ float red2[16 * 8];
  __shared__ float mu_s[16];
  __shared__ float rs_s[16];

  const int lane = threadIdx.x & 31;
  const int wv   = threadIdx.x >> 5;
  const int m0   = blockIdx.x * 16;
  const int mt0  = m0 >> 4;

  v8f acc[8] = {};
  for (int kb = 0; kb < KBLK; ++kb) {
    v16h a = *(const v16h*)(A + (((mt0 * KBLK + kb) * 32) + lane) * 16);
#pragma unroll
    for (int nt = 0; nt < 8; ++nt) {
      int ntile = wv * 8 + nt;
      v16h b = *(const v16h*)(Bw + (((ntile * KBLK + kb) * 32) + lane) * 16);
      acc[nt] = __builtin_amdgcn_wmma_f32_16x16x32_f16(
          false, a, false, b, (short)0, acc[nt], false, false);
    }
  }

  const int rbase = (lane >> 4) << 3;
  const int cl    = lane & 15;
#pragma unroll
  for (int nt = 0; nt < 8; ++nt) {
    const int n = wv * 128 + nt * 16 + cl;
#pragma unroll
    for (int v = 0; v < 8; ++v) {
      const int r = rbase + v;
      float val = acc[nt][v] + b2[n] + xres[(m0 + r) * EDIM + n];  // + residual
      ylds[r * 512 + n] = val;
    }
  }
  __syncthreads();

  // Row statistics: 8 threads per row, 64 elements each.
  {
    const int row = threadIdx.x >> 3;
    const int seg = threadIdx.x & 7;
    float s1 = 0.f, s2 = 0.f;
    const int e0 = seg * 64;
    for (int e = e0; e < e0 + 64; ++e) {
      float v = ylds[row * 512 + e];
      s1 += v; s2 += v * v;
    }
    red1[row * 8 + seg] = s1;
    red2[row * 8 + seg] = s2;
  }
  __syncthreads();
  if (threadIdx.x < 16) {
    float a1 = 0.f, a2 = 0.f;
#pragma unroll
    for (int j = 0; j < 8; ++j) { a1 += red1[threadIdx.x * 8 + j]; a2 += red2[threadIdx.x * 8 + j]; }
    float mu  = a1 * (1.0f / 512.0f);
    float var = a2 * (1.0f / 512.0f) - mu * mu;
    mu_s[threadIdx.x] = mu;
    rs_s[threadIdx.x] = rsqrtf(var + 1e-5f);
  }
  __syncthreads();

  for (int i = threadIdx.x; i < 16 * 512; i += 128) {
    int r = i >> 9;
    int e = i & 511;
    out[(m0 + r) * EDIM + e] = (ylds[i] - mu_s[r]) * rs_s[r] * gamma[e] + beta[e];
  }
}

// ---------------------------------------------------------------------------
extern "C" void kernel_launch(void* const* d_in, const int* in_sizes, int n_in,
                              void* d_out, int out_size, void* d_ws, size_t ws_size,
                              hipStream_t stream) {
  (void)in_sizes; (void)n_in; (void)out_size; (void)ws_size;

  const float*         x     = (const float*)d_in[0];
  const unsigned char* mask  = (const unsigned char*)d_in[1];  // jnp bool: 1 byte/elem
  const float*         W1    = (const float*)d_in[2];
  const float*         b1    = (const float*)d_in[3];
  const float*         Ww    = (const float*)d_in[4];
  const float*         bw    = (const float*)d_in[5];
  const float*         W2    = (const float*)d_in[6];
  const float*         b2    = (const float*)d_in[7];
  const float*         gamma = (const float*)d_in[8];
  const float*         beta  = (const float*)d_in[9];
  float*               out   = (float*)d_out;

  char* ws = (char*)d_ws;
  size_t off = 0;
  auto alloc = [&](size_t bytes) -> void* {
    void* p = ws + off;
    off += (bytes + 255) & ~(size_t)255;
    return p;
  };
  _Float16* xf     = (_Float16*)alloc((size_t)MTOT * KD * 2);      // x in A-frag f16
  _Float16* hf     = (_Float16*)alloc((size_t)MTOT * KD * 2);      // h in A-frag f16
  _Float16* of     = (_Float16*)alloc((size_t)MTOT * KD * 2);      // conv out, A-frag f16
  float*    logits = (float*)alloc((size_t)MTOT * NWPAD * 4);      // logits/softmax weights
  _Float16* W1f    = (_Float16*)alloc((size_t)KD * 512 * 2);
  _Float16* Wwf    = (_Float16*)alloc((size_t)KD * NWPAD * 2);
  _Float16* W2f    = (_Float16*)alloc((size_t)KD * 512 * 2);

  // Weight / input conversion to WMMA fragment order
  cvt_w_frag<<<(512 * 128) / 256, 256, 0, stream>>>(W1, W1f, 512);
  cvt_w_frag<<<(NWPAD * 128) / 256, 256, 0, stream>>>(Ww, Wwf, 248);
  cvt_w_frag<<<(512 * 128) / 256, 256, 0, stream>>>(W2, W2f, 512);
  cvt_x_frag<<<(MTOT * 128) / 256, 256, 0, stream>>>(x, xf);

  // h = mask ? 0 : x@W1 + b1
  gemm_wmma<0><<<dim3(512 / 128, MTOT / 64), 128, 0, stream>>>(xf, W1f, b1, mask, hf, 512);
  // logits = h@Ww + bw   (N padded 248 -> 256)
  gemm_wmma<1><<<dim3(NWPAD / 128, MTOT / 64), 128, 0, stream>>>(hf, Wwf, bw, nullptr, logits, 248);
  // softmax over K=31 per (row, head)
  softmax31<<<(MTOT * HEADS) / 256, 256, 0, stream>>>(logits);
  // dynamic depthwise conv
  dynconv<<<(MTOT * 128) / 256, 256, 0, stream>>>(hf, logits, of);
  // y = LN(conv@W2 + b2 + x)
  gemm2_ln<<<MTOT / 16, 128, 0, stream>>>(of, W2f, b2, x, gamma, beta, out);
}
